// SwinTransformerBlock_39092792328435
// MI455X (gfx1250) — matrix operand order repaired and checked by reading, not compiled, not run on hardware
//
#include <hip/hip_runtime.h>
#include <math.h>

// ---------------------------------------------------------------------------
// Swin Transformer block for MI455X (gfx1250), wave32 + WMMA bf16 16x16x32.
// B=8, H=W=256, DIM=96, HEADS=4, HD=24, WS=8, N=64, NW=1024 -> 8192 windows.
// B-matrix operands live in LDS pre-swizzled into the WMMA fragment layout
// (two contiguous ds_load_b128 per fragment). Fragment loads for a whole
// k-chain are issued before the WMMA chain so LDS latency overlaps the
// matrix pipe instead of draining dscnt between k-steps.
// ---------------------------------------------------------------------------

typedef __attribute__((ext_vector_type(16))) __bf16          v16bf;
typedef __attribute__((ext_vector_type(8)))  float           v8f;
typedef __attribute__((ext_vector_type(4)))  unsigned int    v4u;

struct Frag32 { v4u lo, hi; };   // 32 bytes = 16 bf16

#define DIMC   96
#define HEADS  4
#define HD     24
#define HDP    32              // head dim padded to WMMA K
#define NWIN   8192            // B * (H/WS) * (W/WS)
#define SCALEQ 0.2041241452319315f   // 24^-0.5

// float -> bf16 bits, round-to-nearest-even
static __device__ inline unsigned short f2bf(float f) {
  unsigned int u = __builtin_bit_cast(unsigned int, f);
  unsigned int r = u + 0x7FFFu + ((u >> 16) & 1u);
  return (unsigned short)(r >> 16);
}
static __device__ inline unsigned int f2bf2(float a, float b) {
  return (unsigned int)f2bf(a) | ((unsigned int)f2bf(b) << 16);
}
// exact j/24 for j in [0,96): head=(j*171)>>12, rem=j-head*24
static __device__ inline void div24(int j, int& q, int& r) {
  q = (j * 171) >> 12;
  r = j - q * 24;
}

static __device__ inline v8f wmma_bf16(v16bf a, v16bf b, v8f c) {
  return __builtin_amdgcn_wmma_f32_16x16x32_bf16(false, a, false, b, (short)0, c,
                                                 false, false);
}

// A fragment: 16x32 bf16 tile, row-major (ld elems/row, 16B-aligned rows).
static __device__ inline v16bf ld_a(const unsigned short* base, int ld, int lane) {
  const unsigned short* p = base + (lane & 15) * ld + ((lane >> 4) << 4);
  Frag32 f;
  f.lo = *(const v4u*)(p);        // K = kb .. kb+7
  f.hi = *(const v4u*)(p + 8);    // K = kb+8 .. kb+15
  return __builtin_bit_cast(v16bf, f);
}

// B fragment from packed tiles: tile t stored as [lane][16 bf16] contiguous.
// Element r of lane L = B[kt*32 + ((L>>4)<<4) + r][nt*16 + (L&15)].
static __device__ inline v16bf ld_bp(const unsigned short* packed, int tile,
                                     int lane) {
  const unsigned short* p = packed + (((unsigned)tile * 32 + lane) << 4);
  Frag32 f;
  f.lo = *(const v4u*)(p);
  f.hi = *(const v4u*)(p + 8);
  return __builtin_bit_cast(v16bf, f);
}

// B fragment for Q@K^T: B[k][n] = M[n][k], M row-major (ld elems/row).
static __device__ inline v16bf ld_bT(const unsigned short* base, int ld, int lane) {
  const unsigned short* p = base + (lane & 15) * ld + ((lane >> 4) << 4);
  Frag32 f;
  f.lo = *(const v4u*)(p);
  f.hi = *(const v4u*)(p + 8);
  return __builtin_bit_cast(v16bf, f);
}

// 3-k-step GEMM with all fragment loads hoisted before the WMMA chain.
static __device__ inline v8f gemm3(const unsigned short* A, int lda, int astep,
                                   const unsigned short* Bp, int btile0,
                                   int btilestep, int lane) {
  v16bf a[3], b[3];
#pragma unroll
  for (int kt = 0; kt < 3; ++kt) {
    a[kt] = ld_a(A + kt * astep, lda, lane);
    b[kt] = ld_bp(Bp, btile0 + kt * btilestep, lane);
  }
  v8f acc = {};
#pragma unroll
  for (int kt = 0; kt < 3; ++kt) acc = wmma_bf16(a[kt], b[kt], acc);
  return acc;
}

// Stage a row-major f32 matrix (ktot x ntot) into packed bf16 B-tiles in LDS.
static __device__ inline void stage_packed(unsigned short* dst,
                                           const float* __restrict__ src,
                                           int ntot, int ntiles, int total,
                                           int tid) {
  for (int i = tid; i < total; i += 256) {
    int e    = i & 15;
    int ln   = (i >> 4) & 31;
    int tile = i >> 9;
    int kt = tile / ntiles, nt = tile - kt * ntiles;
    int k = kt * 32 + ((ln >> 4) << 4) + e;
    int n = nt * 16 + (ln & 15);
    dst[i] = f2bf(src[k * ntot + n]);
  }
}

// shifted-image region label for the attention mask (H=256, WS=8, SHIFT=4)
static __device__ inline int region(int p) {
  return (p < 248) ? 0 : ((p < 252) ? 1 : 2);
}

// ---------------------------------------------------------------------------
// Kernel 1: LN1 + roll(-4,-4) + window partition -> bf16 windows in workspace
// ---------------------------------------------------------------------------
__global__ __launch_bounds__(256) void ln1_shift_window(
    const float* __restrict__ x, const float* __restrict__ g,
    const float* __restrict__ bb, unsigned short* __restrict__ yw) {
  int t = blockIdx.x * 256 + threadIdx.x;       // token id, 524288 total
  int b   = t >> 16;
  int win = (t >> 6) & 1023;
  int n   = t & 63;
  int wh = win >> 5, ww = win & 31;
  int hs = wh * 8 + (n >> 3);
  int wsx = ww * 8 + (n & 7);
  int h = (hs + 4) & 255;                       // y[i] = x[i+SHIFT]
  int w = (wsx + 4) & 255;
  const float* src = x + (((size_t)b * 256 + h) * 256 + w) * DIMC;

  float v[DIMC];
  float mean = 0.f;
#pragma unroll
  for (int c = 0; c < DIMC; ++c) { v[c] = src[c]; mean += v[c]; }
  mean *= (1.f / DIMC);
  float var = 0.f;
#pragma unroll
  for (int c = 0; c < DIMC; ++c) { float d = v[c] - mean; var += d * d; }
  var *= (1.f / DIMC);
  float inv = rsqrtf(var + 1e-5f);

  unsigned int* dst = (unsigned int*)(yw + (size_t)t * DIMC);
#pragma unroll
  for (int c = 0; c < DIMC; c += 2)
    dst[c >> 1] = f2bf2((v[c] - mean) * inv * g[c] + bb[c],
                        (v[c + 1] - mean) * inv * g[c + 1] + bb[c + 1]);
}

// ---------------------------------------------------------------------------
// Kernel 2: per-window QKV GEMM + masked/biased softmax attention + proj +
//           window-reverse + roll(+4,+4) + shortcut add -> d_out (= x1)
// Dynamic LDS (240 KB):
//   Wq packed[27648]us @0      Wp packed[9216]us @55296   Xw[6144]us @73728
//   Qs[8192]us @86016          Ks[8192]us @102400         Vp packed[8192]us @118784
//   S[16384]f32 @135168        P[16384]us @200704         O[6144]us @233472
// ---------------------------------------------------------------------------
__global__ __launch_bounds__(256) void win_attn(
    const unsigned short* __restrict__ yw, const float* __restrict__ x,
    const float* __restrict__ qkv_w, const float* __restrict__ qkv_b,
    const float* __restrict__ bias_tab, const float* __restrict__ proj_w,
    const float* __restrict__ proj_b, float* __restrict__ out) {
  extern __shared__ unsigned char smem[];
  unsigned short* Wq = (unsigned short*)(smem);            // packed 3x18 tiles
  unsigned short* Wp = (unsigned short*)(smem + 55296);    // packed 3x6 tiles
  unsigned short* Xw = (unsigned short*)(smem + 73728);
  unsigned short* Qs = (unsigned short*)(smem + 86016);
  unsigned short* Ks = (unsigned short*)(smem + 102400);
  unsigned short* Vp = (unsigned short*)(smem + 118784);   // packed per head 2x2
  float*          S  = (float*)(smem + 135168);
  unsigned short* P  = (unsigned short*)(smem + 200704);
  unsigned short* O  = (unsigned short*)(smem + 233472);

  const int tid  = threadIdx.x;
  const int lane = tid & 31;
  const int wave = __builtin_amdgcn_readfirstlane(tid >> 5);  // SGPR-uniform
  const int win  = blockIdx.x;
  const int b    = win >> 10;
  const int wl   = win & 1023;
  const int wh   = wl >> 5, ww = wl & 31;

  __builtin_prefetch(qkv_w, 0, 1);
  __builtin_prefetch(proj_w, 0, 1);

  // ---- stage: weights pre-packed to fragment layout, window copy, zero QKV ----
  stage_packed(Wq, qkv_w, 288, 18, 27648, tid);
  stage_packed(Wp, proj_w, 96, 6, 9216, tid);
  {
    const v4u* s4 = (const v4u*)(yw + (size_t)win * 6144);
    v4u* d4 = (v4u*)Xw;
    for (int i = tid; i < 768; i += 256) d4[i] = s4[i];
    v4u z = {0u, 0u, 0u, 0u};
    v4u* q4 = (v4u*)Qs;                    // Qs,Ks,Vp contiguous: 49152 B
    for (int i = tid; i < 3072; i += 256) q4[i] = z;
  }
  __syncthreads();

  // ---- QKV = Xw(64x96) @ Wq(96x288) + b; scatter into Q/K (padded) & packed V
  for (int t = wave; t < 72; t += 8) {          // t, nt, m0 all scalar
    const int m0 = (t & 3) * 16, nt = t >> 2;   // 4 x 18 tiles
    v8f acc = gemm3(Xw + m0 * 96, 96, 32, Wq, nt, 18, lane);
    const int rb = (lane >> 4) << 3;
    const int j  = nt * 16 + (lane & 15);
    float bj = qkv_b[j];
    if (nt < 6) {                               // Q segment (scalar branch)
      int head, d; div24(j, head, d);
      unsigned short* q0 = Qs + head * 2048 + (m0 + rb) * HDP + d;
#pragma unroll
      for (int r = 0; r < 8; ++r)
        q0[r * HDP] = f2bf((acc[r] + bj) * SCALEQ);
    } else if (nt < 12) {                       // K segment
      int head, d; div24(j - 96, head, d);
      unsigned short* k0 = Ks + head * 2048 + (m0 + rb) * HDP + d;
#pragma unroll
      for (int r = 0; r < 8; ++r)
        k0[r * HDP] = f2bf(acc[r] + bj);
    } else {                                    // V segment -> packed B tiles
      int head, d; div24(j - 192, head, d);
      int row0 = m0 + rb;                       // rows row0..row0+7 (same 16-blk)
      int vt  = (row0 >> 5) * 2 + (d >> 4);     // kt*2 + nt
      int lnp = (((row0 >> 4) & 1) << 4) + (d & 15);
      unsigned short* v0 = Vp + head * 2048 + ((vt * 32 + lnp) << 4) + (row0 & 15);
#pragma unroll
      for (int r = 0; r < 8; ++r)
        v0[r] = f2bf(acc[r] + bj);              // contiguous -> b128 store
    }
  }
  __syncthreads();

  // ---- S = Q @ K^T per head (K = 24 padded to 32, single k-step) ----
  for (int t = wave; t < 64; t += 8) {
    int head = t >> 4, m0 = ((t >> 2) & 3) * 16, n0 = (t & 3) * 16;
    v16bf a = ld_a(Qs + head * 2048 + m0 * HDP, HDP, lane);
    v16bf bm = ld_bT(Ks + head * 2048 + n0 * HDP, HDP, lane);
    v8f acc = {};
    acc = wmma_bf16(a, bm, acc);
    float* s0 = S + head * 4096 + (m0 + ((lane >> 4) << 3)) * 64 + n0 + (lane & 15);
#pragma unroll
    for (int r = 0; r < 8; ++r) s0[r * 64] = acc[r];
  }
  __syncthreads();

  // ---- softmax row-per-thread: +mask, +clip(bias,±5), clip(±10) ----
  {
    int head = tid >> 6, row = tid & 63;
    int ri = row >> 3, ci = row & 7;
    int cr = region(wh * 8 + ri) * 3 + region(ww * 8 + ci);
    float* Srow = S + head * 4096 + row * 64;
    float sv[64];
    float mx = -1e30f;
#pragma unroll
    for (int m = 0; m < 64; ++m) {
      int rm = m >> 3, cm = m & 7;
      int cc = region(wh * 8 + rm) * 3 + region(ww * 8 + cm);
      float msk = (cc != cr) ? -1e9f : 0.f;
      int bidx = (ri - rm + 7) * 15 + (ci - cm + 7);
      float bi = bias_tab[(bidx << 2) + head];
      bi = fminf(fmaxf(bi, -5.f), 5.f);
      float val = fminf(fmaxf(Srow[m] + msk + bi, -10.f), 10.f);
      sv[m] = val;
      mx = fmaxf(mx, val);
    }
    float sum = 0.f;
#pragma unroll
    for (int m = 0; m < 64; ++m) { sv[m] = __expf(sv[m] - mx); sum += sv[m]; }
    float rs = 1.f / sum;
    unsigned int* Prow = (unsigned int*)(P + head * 4096 + row * 64);
#pragma unroll
    for (int m = 0; m < 32; ++m)
      Prow[m] = f2bf2(sv[2 * m] * rs, sv[2 * m + 1] * rs);
  }
  __syncthreads();

  // ---- O = P(64x64) @ V(64x32 padded); keep first 24 cols per head ----
  for (int t = wave; t < 32; t += 8) {
    int head = t >> 3, m0 = ((t >> 1) & 3) * 16, nt = t & 1;
    v16bf a[2], bm[2];
#pragma unroll
    for (int kt = 0; kt < 2; ++kt) {
      a[kt]  = ld_a(P + head * 4096 + m0 * 64 + kt * 32, 64, lane);
      bm[kt] = ld_bp(Vp + head * 2048, kt * 2 + nt, lane);
    }
    v8f acc = {};
#pragma unroll
    for (int kt = 0; kt < 2; ++kt) acc = wmma_bf16(a[kt], bm[kt], acc);
    int d  = nt * 16 + (lane & 15);
    unsigned short* o0 = O + (m0 + ((lane >> 4) << 3)) * 96 + head * 24 + d;
    if (d < 24) {                               // only nt==1 partially masks
#pragma unroll
      for (int r = 0; r < 8; ++r) o0[r * 96] = f2bf(acc[r]);
    }
  }
  __syncthreads();

  // ---- proj + window reverse + roll(+4,+4) + shortcut -> out (x1) ----
  for (int t = wave; t < 24; t += 8) {
    const int m0 = (t & 3) * 16, nt = t >> 2;   // 4 x 6 tiles
    v8f acc = gemm3(O + m0 * 96, 96, 32, Wp, nt, 6, lane);
    int j  = nt * 16 + (lane & 15);
    int rb = (lane >> 4) << 3;
    float pb = proj_b[j];
#pragma unroll
    for (int r = 0; r < 8; ++r) {
      int row = m0 + rb + r;
      int hs = wh * 8 + (row >> 3), wsx = ww * 8 + (row & 7);
      int h = (hs + 4) & 255, w = (wsx + 4) & 255;
      size_t idx = (((size_t)b * 256 + h) * 256 + w) * DIMC + j;
      out[idx] = x[idx] + acc[r] + pb;
    }
  }
}

// ---------------------------------------------------------------------------
// Kernel 3: in-place per-64-row: LN2 -> GEMM(96->384) -> exact GELU ->
//           GEMM(384->96) -> residual add.   Dynamic LDS: 204 KB
//   Xl[6144]us @0   W1 packed[36864]us @12288   G[24576]us @86016
//   W2 packed[36864]us @135168
// ---------------------------------------------------------------------------
__global__ __launch_bounds__(256) void mlp_block(
    float* __restrict__ out, const float* __restrict__ g2,
    const float* __restrict__ b2, const float* __restrict__ w1,
    const float* __restrict__ mb1, const float* __restrict__ w2,
    const float* __restrict__ mb2) {
  extern __shared__ unsigned char smem[];
  unsigned short* Xl = (unsigned short*)(smem);
  unsigned short* W1 = (unsigned short*)(smem + 12288);    // packed 3x24 tiles
  unsigned short* G  = (unsigned short*)(smem + 86016);
  unsigned short* W2 = (unsigned short*)(smem + 135168);   // packed 12x6 tiles

  const int tid = threadIdx.x, lane = tid & 31;
  const int wave = __builtin_amdgcn_readfirstlane(tid >> 5);
  const size_t row0 = (size_t)blockIdx.x * 64;

  __builtin_prefetch(w1, 0, 1);
  __builtin_prefetch(w2, 0, 1);

  stage_packed(W1, w1, 384, 24, 36864, tid);
  stage_packed(W2, w2, 96, 6, 36864, tid);
  if (tid < 64) {
    const float* src = out + (row0 + tid) * DIMC;     // x1 rows (own rows only)
    float v[DIMC];
    float mean = 0.f;
#pragma unroll
    for (int c = 0; c < DIMC; ++c) { v[c] = src[c]; mean += v[c]; }
    mean *= (1.f / DIMC);
    float var = 0.f;
#pragma unroll
    for (int c = 0; c < DIMC; ++c) { float d = v[c] - mean; var += d * d; }
    var *= (1.f / DIMC);
    float inv = rsqrtf(var + 1e-5f);
    unsigned int* dst = (unsigned int*)(Xl + tid * DIMC);
#pragma unroll
    for (int c = 0; c < DIMC; c += 2)
      dst[c >> 1] = f2bf2((v[c] - mean) * inv * g2[c] + b2[c],
                          (v[c + 1] - mean) * inv * g2[c + 1] + b2[c + 1]);
  }
  __syncthreads();

  // G = gelu(Xl(64x96) @ W1(96x384) + b1)   4 x 24 tiles
  for (int t = wave; t < 96; t += 8) {
    const int m0 = (t & 3) * 16, nt = t >> 2;
    v8f acc = gemm3(Xl + m0 * 96, 96, 32, W1, nt, 24, lane);
    int j  = nt * 16 + (lane & 15);
    float bj = mb1[j];
    unsigned short* g0 = G + (m0 + ((lane >> 4) << 3)) * 384 + j;
#pragma unroll
    for (int r = 0; r < 8; ++r) {
      float val = acc[r] + bj;
      float ge = 0.5f * val * (1.f + erff(val * 0.70710678118654752f));
      g0[r * 384] = f2bf(ge);
    }
  }
  __syncthreads();

  // out += G(64x384) @ W2(384x96) + b2      4 x 6 tiles, 12 k-steps (4-chunked)
  for (int t = wave; t < 24; t += 8) {
    const int m0 = (t & 3) * 16, nt = t >> 2;
    v8f acc = {};
#pragma unroll
    for (int kc = 0; kc < 12; kc += 4) {        // 4 k-steps in flight per chunk
      v16bf a[4], bm[4];
#pragma unroll
      for (int kt = 0; kt < 4; ++kt) {
        a[kt]  = ld_a(G + m0 * 384 + (kc + kt) * 32, 384, lane);
        bm[kt] = ld_bp(W2, (kc + kt) * 6 + nt, lane);
      }
#pragma unroll
      for (int kt = 0; kt < 4; ++kt) acc = wmma_bf16(a[kt], bm[kt], acc);
    }
    int j  = nt * 16 + (lane & 15);
    float bj = mb2[j];
    float* d0 = out + (row0 + m0 + ((lane >> 4) << 3)) * DIMC + j;
#pragma unroll
    for (int r = 0; r < 8; ++r)
      d0[r * DIMC] = d0[r * DIMC] + acc[r] + bj;
  }
}

// ---------------------------------------------------------------------------
extern "C" void kernel_launch(void* const* d_in, const int* in_sizes, int n_in,
                              void* d_out, int out_size, void* d_ws,
                              size_t ws_size, hipStream_t stream) {
  (void)in_sizes; (void)n_in; (void)out_size; (void)ws_size;
  const float* x    = (const float*)d_in[0];
  const float* n1g  = (const float*)d_in[1];
  const float* n1b  = (const float*)d_in[2];
  const float* qw   = (const float*)d_in[3];
  const float* qb   = (const float*)d_in[4];
  const float* rbt  = (const float*)d_in[5];
  const float* pw   = (const float*)d_in[6];
  const float* pb   = (const float*)d_in[7];
  const float* n2g  = (const float*)d_in[8];
  const float* n2b  = (const float*)d_in[9];
  const float* w1   = (const float*)d_in[10];
  const float* mb1  = (const float*)d_in[11];
  const float* w2   = (const float*)d_in[12];
  const float* mb2  = (const float*)d_in[13];
  float* out = (float*)d_out;
  unsigned short* yw = (unsigned short*)d_ws;    // 8192*64*96 bf16 = 96 MiB

  hipFuncSetAttribute((const void*)win_attn,
                      hipFuncAttributeMaxDynamicSharedMemorySize, 245760);
  hipFuncSetAttribute((const void*)mlp_block,
                      hipFuncAttributeMaxDynamicSharedMemorySize, 208896);

  // 1) LN1 + shift + window partition (524288 tokens)
  ln1_shift_window<<<2048, 256, 0, stream>>>(x, n1g, n1b, yw);
  // 2) per-window attention block -> writes x1 into d_out (full overwrite)
  win_attn<<<NWIN, 256, 245760, stream>>>(yw, x, qw, qb, rbt, pw, pb, out);
  // 3) in-place LN2 + MLP + residual on d_out
  mlp_block<<<NWIN, 256, 208896, stream>>>(out, n2g, n2b, w1, mb1, w2, mb2);
}
